// GraphNetwork_35957466202280
// MI455X (gfx1250) — matrix operand order repaired and checked by reading, not compiled, not run on hardware
//
#include <hip/hip_runtime.h>

#define HID 128
#define TILE_ROWS 128   // rows per block
#define NTHREADS 128    // 4 waves; each wave owns 32 rows (2 x 16-row tiles)

typedef __attribute__((ext_vector_type(16))) _Float16 v16h;
typedef __attribute__((ext_vector_type(8)))  _Float16 v8h;
typedef __attribute__((ext_vector_type(8)))  float    v8f;

// Swizzled weight image offsets (in halfs) inside the f16 weight arena in d_ws.
// Image sizes: 32x128 -> 4096, 128x128 -> 16384, 128x16 -> 2048.
#define E0_OFF 0
#define E1_OFF 4096
#define E2_OFF 20480
#define E3_OFF 36864
#define U0_OFF 38912
#define U1_OFF 43008
#define U2_OFF 59392
#define U3_OFF 75776

// ---- WMMA fragment helpers (wave32; C/D layout per CDNA5 ISA 7.12.2; A/B use contiguous k = 16*(lane>>4) + i) ----

// A: 16x32 f16 from row-major LDS (leading dim multiple of 8, 16B aligned).
// lane holds row (lane&15); halves = K[k0..k0+7], K[k0+8..k0+15], k0 = 16*(lane>>4).
__device__ __forceinline__ v16h load_a_frag(const _Float16* base, int ld, int lane) {
  const int row = lane & 15;
  const int k0  = (lane >> 4) << 4;
  const _Float16* p = base + row * ld + k0;
  v8h lo = *(const v8h*)(p);
  v8h hi = *(const v8h*)(p + 8);
  return __builtin_shufflevector(lo, hi, 0, 1, 2, 3, 4, 5, 6, 7,
                                 8, 9, 10, 11, 12, 13, 14, 15);
}

// B fragments pre-swizzled: image layout [frag][lane][16 halfs] contiguous.
__device__ __forceinline__ v16h load_b_frag_swz(const _Float16* sW, int frag, int lane) {
  return *(const v16h*)(sW + ((frag << 5) + lane) * 16);
}

__device__ __forceinline__ v8f wmma_f16(v16h a, v16h b, v8f c) {
  return __builtin_amdgcn_wmma_f32_16x16x32_f16(false, a, false, b, (short)0, c,
                                                false, false);
}

// C/D f32 16x16: lane holds col = lane&15, rows (lane>>4)*8 + i for VGPR i.
__device__ __forceinline__ void epilogue_relu_store(const v8f* acc,
                                                    const float* __restrict__ bias,
                                                    _Float16* dst, int lane) {
  const int col = lane & 15;
  const int rb  = (lane >> 4) << 3;
#pragma unroll
  for (int n = 0; n < 8; ++n) {
    const float bv = bias[n * 16 + col];
#pragma unroll
    for (int i = 0; i < 8; ++i) {
      float v = acc[n][i] + bv;
      v = v > 0.f ? v : 0.f;
      dst[(rb + i) * HID + n * 16 + col] = (_Float16)v;
    }
  }
}

// Contiguous f16 copy global -> LDS (b128 granularity).
__device__ __forceinline__ void stage(const _Float16* __restrict__ src,
                                      _Float16* dst, int halfs, int tid) {
  const int vecs = halfs >> 3;
  for (int i = tid; i < vecs; i += NTHREADS)
    ((v8h*)dst)[i] = ((const v8h*)src)[i];
}

__device__ __forceinline__ v8f vzero8() {
  v8f z = {0.f, 0.f, 0.f, 0.f, 0.f, 0.f, 0.f, 0.f};
  return z;
}

// One 128->128 ReLU layer for this wave's TWO 16-row tiles (in-place in LDS).
// Each B fragment read from LDS feeds two WMMAs (register row-blocking).
__device__ __forceinline__ void dense128_relu_x2(_Float16* tile0, _Float16* tile1,
                                                 const _Float16* sW,
                                                 const float* __restrict__ bias,
                                                 int lane) {
  v8f acc0[8], acc1[8];
#pragma unroll
  for (int n = 0; n < 8; ++n) { acc0[n] = vzero8(); acc1[n] = vzero8(); }
#pragma unroll
  for (int k = 0; k < 4; ++k) {
    v16h a0 = load_a_frag(tile0 + k * 32, HID, lane);
    v16h a1 = load_a_frag(tile1 + k * 32, HID, lane);
#pragma unroll
    for (int n = 0; n < 8; ++n) {
      v16h b = load_b_frag_swz(sW, k * 8 + n, lane);
      acc0[n] = wmma_f16(a0, b, acc0[n]);
      acc1[n] = wmma_f16(a1, b, acc1[n]);
    }
  }
  epilogue_relu_store(acc0, bias, tile0, lane);
  epilogue_relu_store(acc1, bias, tile1, lane);
}

// Head: [32,128] x [128,16(pad)] -> two C fragments.
__device__ __forceinline__ void dense_head_x2(const _Float16* tile0,
                                              const _Float16* tile1,
                                              const _Float16* sW, int lane,
                                              v8f* m0, v8f* m1) {
  v8f a = vzero8(), b2 = vzero8();
#pragma unroll
  for (int k = 0; k < 4; ++k) {
    v16h a0 = load_a_frag(tile0 + k * 32, HID, lane);
    v16h a1 = load_a_frag(tile1 + k * 32, HID, lane);
    v16h b = load_b_frag_swz(sW, k, lane);   // nt_n == 1 -> frag == kt
    a  = wmma_f16(a0, b, a);
    b2 = wmma_f16(a1, b, b2);
  }
  *m0 = a;
  *m1 = b2;
}

// =====================================================================
// Kernel 0: zero the pooled accumulator
// =====================================================================
__global__ void zero_f32(float* __restrict__ p, int n) {
  int i = blockIdx.x * blockDim.x + threadIdx.x;
  if (i < n) p[i] = 0.f;
}

// =====================================================================
// Kernel P: swizzle all 8 weight matrices into fragment-native f16 images.
//   img[(f*32+l)*16+i] = W[(kt*32 + 16*(l>>4) + i)*N + nt*16 + (l&15)]  (0-padded)
// =====================================================================
__device__ __forceinline__ void swizzle_one(const float* __restrict__ W,
                                            int K, int N, int Kp, int Np,
                                            _Float16* __restrict__ dst) {
  const int nt_n  = Np >> 4;
  const int total = (Kp >> 5) * nt_n * 32;   // frag-lanes
  for (int fl = threadIdx.x; fl < total; fl += NTHREADS) {
    const int l  = fl & 31;
    const int f  = fl >> 5;
    const int kt = f / nt_n;
    const int nt = f - kt * nt_n;
    const int nn = nt * 16 + (l & 15);
    const int kbase = kt * 32 + ((l >> 4) << 4);
    v16h t;
#pragma unroll
    for (int i = 0; i < 16; ++i) {
      const int kk = kbase + i;
      float v = (kk < K && nn < N) ? W[kk * N + nn] : 0.f;
      t[i] = (_Float16)v;
    }
    *(v16h*)(dst + fl * 16) = t;
  }
}

__global__ void __launch_bounds__(NTHREADS) swizzle_weights(
    const float* __restrict__ Wm0, const float* __restrict__ Wm1,
    const float* __restrict__ Wm2, const float* __restrict__ Wm3,
    const float* __restrict__ Wu0, const float* __restrict__ Wu1,
    const float* __restrict__ Wu2, const float* __restrict__ Wu3,
    _Float16* __restrict__ wbase) {
  switch (blockIdx.x) {
    case 0: swizzle_one(Wm0,  10, 128,  32, 128, wbase + E0_OFF); break;
    case 1: swizzle_one(Wm1, 128, 128, 128, 128, wbase + E1_OFF); break;
    case 2: swizzle_one(Wm2, 128, 128, 128, 128, wbase + E2_OFF); break;
    case 3: swizzle_one(Wm3, 128,   2, 128,  16, wbase + E3_OFF); break;
    case 4: swizzle_one(Wu0,   7, 128,  32, 128, wbase + U0_OFF); break;
    case 5: swizzle_one(Wu1, 128, 128, 128, 128, wbase + U1_OFF); break;
    case 6: swizzle_one(Wu2, 128, 128, 128, 128, wbase + U2_OFF); break;
    case 7: swizzle_one(Wu3, 128,   2, 128,  16, wbase + U3_OFF); break;
  }
}

// =====================================================================
// Kernel 1: edge MLP + atomic scatter-add into pooled[N,2]
// =====================================================================
__global__ void __launch_bounds__(NTHREADS) edge_mlp_scatter(
    const float* __restrict__ nodes,
    const int*   __restrict__ senders,
    const int*   __restrict__ receivers,
    const _Float16* __restrict__ wE,          // swizzled edge-MLP weights
    const float* __restrict__ bm0, const float* __restrict__ bm1,
    const float* __restrict__ bm2, const float* __restrict__ bm3,
    float* __restrict__ pooled,
    int num_edges) {
  __shared__ __align__(32) _Float16 sW[HID * HID];       // 32 KB weight stage
  __shared__ __align__(32) _Float16 sH[TILE_ROWS * HID]; // 32 KB activations

  const int tid  = threadIdx.x;
  const int lane = tid & 31;
  const int wave = tid >> 5;
  const int e0   = (int)blockIdx.x * TILE_ROWS;
  _Float16* myH0 = sH + (wave * 32) * HID;
  _Float16* myH1 = myH0 + 16 * HID;

  // ---- stage swizzled W0 + gather x0[128][32] (at sW+8192, overwritten later)
  stage(wE + E0_OFF, sW, 4096, tid);
  {
    int r = tid;                 // one row per thread
    int e = e0 + r;
    if (e >= num_edges) e = num_edges - 1;
    int s  = senders[e];
    int rc = receivers[e];
    v16h t, z;
#pragma unroll
    for (int j = 0; j < 16; ++j) z[j] = (_Float16)0.f;
#pragma unroll
    for (int j = 0; j < 16; ++j) {
      float v = 0.f;
      if (j < 5)       v = nodes[s * 5 + j];
      else if (j < 10) v = nodes[rc * 5 + (j - 5)];
      t[j] = (_Float16)v;
    }
    *(v16h*)(sW + 8192 + r * 32)      = t;
    *(v16h*)(sW + 8192 + r * 32 + 16) = z;
  }
  __syncthreads();

  // ---- layer 0: [32,32(pad)] x [32,128] ----
  {
    v8f acc0[8], acc1[8];
#pragma unroll
    for (int n = 0; n < 8; ++n) { acc0[n] = vzero8(); acc1[n] = vzero8(); }
    v16h a0 = load_a_frag(sW + 8192 + (wave * 32) * 32, 32, lane);
    v16h a1 = load_a_frag(sW + 8192 + (wave * 32 + 16) * 32, 32, lane);
#pragma unroll
    for (int n = 0; n < 8; ++n) {
      v16h b = load_b_frag_swz(sW, n, lane);   // kt==0 -> frag == nt
      acc0[n] = wmma_f16(a0, b, acc0[n]);
      acc1[n] = wmma_f16(a1, b, acc1[n]);
    }
    epilogue_relu_store(acc0, bm0, myH0, lane);
    epilogue_relu_store(acc1, bm0, myH1, lane);
  }
  __syncthreads();

  stage(wE + E1_OFF, sW, 16384, tid);
  __syncthreads();
  dense128_relu_x2(myH0, myH1, sW, bm1, lane);
  __syncthreads();

  stage(wE + E2_OFF, sW, 16384, tid);
  __syncthreads();
  dense128_relu_x2(myH0, myH1, sW, bm2, lane);
  __syncthreads();

  stage(wE + E3_OFF, sW, 2048, tid);
  __syncthreads();
  {
    v8f m0, m1;
    dense_head_x2(myH0, myH1, sW, lane, &m0, &m1);
    const int col = lane & 15;
    const int rb  = (lane >> 4) << 3;
    if (col < 2) {
      const float bv = bm3[col];
#pragma unroll
      for (int i = 0; i < 8; ++i) {
        int e = e0 + wave * 32 + rb + i;
        if (e < num_edges) {
          atomicAdd(pooled + receivers[e] * 2 + col, m0[i] + bv);
        }
      }
#pragma unroll
      for (int i = 0; i < 8; ++i) {
        int e = e0 + wave * 32 + 16 + rb + i;
        if (e < num_edges) {
          atomicAdd(pooled + receivers[e] * 2 + col, m1[i] + bv);
        }
      }
    }
  }
}

// =====================================================================
// Kernel 2: node MLP + Euler update of velocity columns
// =====================================================================
__global__ void __launch_bounds__(NTHREADS) node_mlp_update(
    const float* __restrict__ nodes,
    const float* __restrict__ pooled,
    const _Float16* __restrict__ wU,          // swizzled node-MLP weights (U0 base)
    const float* __restrict__ bu0, const float* __restrict__ bu1,
    const float* __restrict__ bu2, const float* __restrict__ bu3,
    float* __restrict__ out,
    int num_nodes) {
  __shared__ __align__(32) _Float16 sW[HID * HID];
  __shared__ __align__(32) _Float16 sH[TILE_ROWS * HID];

  const int tid  = threadIdx.x;
  const int lane = tid & 31;
  const int wave = tid >> 5;
  const int n0   = (int)blockIdx.x * TILE_ROWS;
  _Float16* myH0 = sH + (wave * 32) * HID;
  _Float16* myH1 = myH0 + 16 * HID;

  stage(wU, sW, 4096, tid);
  {
    int r = tid;
    int n = n0 + r;
    if (n >= num_nodes) n = num_nodes - 1;
    v16h t, z;
#pragma unroll
    for (int j = 0; j < 16; ++j) z[j] = (_Float16)0.f;
#pragma unroll
    for (int j = 0; j < 16; ++j) {
      float v = 0.f;
      if (j < 5)      v = nodes[n * 5 + j];
      else if (j < 7) v = pooled[n * 2 + (j - 5)];
      t[j] = (_Float16)v;
    }
    *(v16h*)(sW + 8192 + r * 32)      = t;
    *(v16h*)(sW + 8192 + r * 32 + 16) = z;
  }
  __syncthreads();

  {
    v8f acc0[8], acc1[8];
#pragma unroll
    for (int n = 0; n < 8; ++n) { acc0[n] = vzero8(); acc1[n] = vzero8(); }
    v16h a0 = load_a_frag(sW + 8192 + (wave * 32) * 32, 32, lane);
    v16h a1 = load_a_frag(sW + 8192 + (wave * 32 + 16) * 32, 32, lane);
#pragma unroll
    for (int n = 0; n < 8; ++n) {
      v16h b = load_b_frag_swz(sW, n, lane);
      acc0[n] = wmma_f16(a0, b, acc0[n]);
      acc1[n] = wmma_f16(a1, b, acc1[n]);
    }
    epilogue_relu_store(acc0, bu0, myH0, lane);
    epilogue_relu_store(acc1, bu0, myH1, lane);
  }
  __syncthreads();

  stage(wU + (U1_OFF - U0_OFF), sW, 16384, tid);
  __syncthreads();
  dense128_relu_x2(myH0, myH1, sW, bu1, lane);
  __syncthreads();

  stage(wU + (U2_OFF - U0_OFF), sW, 16384, tid);
  __syncthreads();
  dense128_relu_x2(myH0, myH1, sW, bu2, lane);
  __syncthreads();

  stage(wU + (U3_OFF - U0_OFF), sW, 2048, tid);
  __syncthreads();

  v8f m0, m1;
  dense_head_x2(myH0, myH1, sW, lane, &m0, &m1);
  __syncthreads();  // everyone done reading sH before it is reused as sD

  float* sD = (float*)sH;  // delta stage [128][2]
  {
    const int col = lane & 15;
    const int rb  = (lane >> 4) << 3;
    if (col < 2) {
      const float bv = bu3[col];
#pragma unroll
      for (int i = 0; i < 8; ++i) {
        int r = wave * 32 + rb + i;
        sD[r * 2 + col] = m0[i] + bv;
      }
#pragma unroll
      for (int i = 0; i < 8; ++i) {
        int r = wave * 32 + 16 + rb + i;
        sD[r * 2 + col] = m1[i] + bv;
      }
    }
  }
  __syncthreads();

  for (int r = tid; r < TILE_ROWS; r += NTHREADS) {
    int n = n0 + r;
    if (n < num_nodes) {
      out[n * 5 + 0] = nodes[n * 5 + 0];
      out[n * 5 + 1] = nodes[n * 5 + 1];
      out[n * 5 + 2] = nodes[n * 5 + 2] + sD[r * 2 + 0];
      out[n * 5 + 3] = nodes[n * 5 + 3] + sD[r * 2 + 1];
      out[n * 5 + 4] = nodes[n * 5 + 4];
    }
  }
}

// =====================================================================
extern "C" void kernel_launch(void* const* d_in, const int* in_sizes, int n_in,
                              void* d_out, int out_size, void* d_ws, size_t ws_size,
                              hipStream_t stream) {
  const float* nodes      = (const float*)d_in[0];
  const int*   edge_index = (const int*)d_in[1];
  const float* Wm0 = (const float*)d_in[2];
  const float* bm0 = (const float*)d_in[3];
  const float* Wm1 = (const float*)d_in[4];
  const float* bm1 = (const float*)d_in[5];
  const float* Wm2 = (const float*)d_in[6];
  const float* bm2 = (const float*)d_in[7];
  const float* Wm3 = (const float*)d_in[8];
  const float* bm3 = (const float*)d_in[9];
  const float* Wu0 = (const float*)d_in[10];
  const float* bu0 = (const float*)d_in[11];
  const float* Wu1 = (const float*)d_in[12];
  const float* bu1 = (const float*)d_in[13];
  const float* Wu2 = (const float*)d_in[14];
  const float* bu2 = (const float*)d_in[15];
  const float* Wu3 = (const float*)d_in[16];
  const float* bu3 = (const float*)d_in[17];

  const int num_nodes = in_sizes[0] / 5;
  const int num_edges = in_sizes[1] / 2;
  const int* senders   = edge_index;              // edge_index[0, :]
  const int* receivers = edge_index + num_edges;  // edge_index[1, :]

  // d_ws layout: [pooled f32 N*2][256B align][f16 swizzled weight arena]
  float* pooled = (float*)d_ws;
  size_t pooled_bytes = (size_t)num_nodes * 2 * sizeof(float);
  size_t warena_off = (pooled_bytes + 255) & ~(size_t)255;
  _Float16* wbase = (_Float16*)((char*)d_ws + warena_off);

  float* out = (float*)d_out;

  const int pn = num_nodes * 2;
  zero_f32<<<(pn + 255) / 256, 256, 0, stream>>>(pooled, pn);

  swizzle_weights<<<8, NTHREADS, 0, stream>>>(Wm0, Wm1, Wm2, Wm3,
                                              Wu0, Wu1, Wu2, Wu3, wbase);

  const int eblocks = (num_edges + TILE_ROWS - 1) / TILE_ROWS;
  edge_mlp_scatter<<<eblocks, NTHREADS, 0, stream>>>(
      nodes, senders, receivers, wbase /* E0_OFF == 0 */,
      bm0, bm1, bm2, bm3, pooled, num_edges);

  const int nblocks = (num_nodes + TILE_ROWS - 1) / TILE_ROWS;
  node_mlp_update<<<nblocks, NTHREADS, 0, stream>>>(
      nodes, pooled, wbase + U0_OFF,
      bu0, bu1, bu2, bu3, out, num_nodes);
}
